// LookAtMappingNetwork_54425825575566
// MI455X (gfx1250) — matrix operand order repaired and checked by reading, not compiled, not run on hardware
//
#include <hip/hip_runtime.h>
#include <cstdint>
#include <cstddef>
#include <cmath>

// ---------------------------------------------------------------------------
// MI455X (gfx1250) implementation of the LookAt GraphNet mapping network.
// Heavy math: V_WMMA_F32_16X16X32_BF16 (wave32 WMMA).
// Tile staging: GLOBAL_LOAD_ASYNC_TO_LDS_B128 (ASYNCcnt) when the toolchain
// exposes the builtin, else pipelined global->reg->ds_store.
// ---------------------------------------------------------------------------

typedef __attribute__((ext_vector_type(16))) __bf16 v16bf;
typedef __attribute__((ext_vector_type(8)))  float  v8f;
typedef unsigned short u16;

#define NV     250
#define NB     64
#define MNODE  16000     // NB * NV
#define EDGES  95232     // NB * 1488
#define DDIM   512
#define NUMWS  14
#define LRC    0.01f
#define SQRT2C 1.41421356237309515f

#if __has_builtin(__builtin_amdgcn_global_load_async_to_lds_b128)
#define USE_ASYNC_LDS 1
#else
#define USE_ASYNC_LDS 0
#endif

#if USE_ASYNC_LDS
// Builtin signature (from hipcc diagnostic): param0 = int4 addrspace(1)*,
// param1 = int4 addrspace(3)*, then two imm ints (offset, cpol).
typedef int v4i __attribute__((vector_size(16)));
typedef __attribute__((address_space(1))) v4i g4i;
typedef __attribute__((address_space(3))) v4i l4i;
__device__ __forceinline__ g4i* to_g4(const void* p) { return (g4i*)(uintptr_t)p; }
__device__ __forceinline__ l4i* to_l4(void* p)       { return (l4i*)(uintptr_t)p; }
#endif

__device__ __forceinline__ void wait_async0() {
#if __has_builtin(__builtin_amdgcn_s_wait_asynccnt)
  __builtin_amdgcn_s_wait_asynccnt(0);
#else
  asm volatile("s_wait_asynccnt 0" ::: "memory");
#endif
}

__device__ __forceinline__ u16 f2b(float f) {
  union { float f; unsigned u; } c; c.f = f;
  unsigned r = (c.u + 0x7FFFu + ((c.u >> 16) & 1u)) >> 16;   // RNE
  return (u16)r;
}
__device__ __forceinline__ float b2f(u16 h) {
  union { unsigned u; float f; } c; c.u = ((unsigned)h) << 16;
  return c.f;
}

// ---------------------------------------------------------------------------
// Fused gather-GEMM:  out = leakyrelu(A @ W^T * gain + bias*LR) * sqrt2
//
// A rows are virtual concatenations of up to 3 bf16 segments.  All segment
// boundaries are multiples of BK=32, so the active segment depends only on
// the (wave-uniform) K-step: scalar select over three row pointers hoisted
// out of the loop.  Tile 128x64, BK=32; 8 waves, wave w owns rows w*16.. and
// all 64 N columns -> 4 WMMAs per K-step.  Double-buffered LDS, one barrier
// per iteration.
// ---------------------------------------------------------------------------
__global__ __launch_bounds__(256)
void gemm_fc(const u16* __restrict__ base0, const int* __restrict__ idx0, int st0,
             const u16* __restrict__ base1, const int* __restrict__ idx1, int st1,
             const u16* __restrict__ base2, const int* __restrict__ idx2, int st2,
             int kb1, int kb2, int Kpad,
             const u16* __restrict__ Wb, const float* __restrict__ bias,
             float gain, u16* __restrict__ out)
{
  __shared__ __align__(16) u16 At[2][128 * 32];
  __shared__ __align__(16) u16 Bt[2][64 * 32];

  const int tid  = threadIdx.x;
  const int m0   = blockIdx.x * 128;
  const int n0   = blockIdx.y * 64;
  const int wave = tid >> 5;      // 0..7 : 16-row A sub-tile
  const int lane = tid & 31;

  // ---- loader precompute (all idx[] loads hoisted) -----------------------
  // A tile: 128x32 halves = 512 16B-chunks; thread t owns chunks t and t+256.
  const int ar0 = tid >> 2;            // rows 0..63
  const int ar1 = ar0 + 64;            // rows 64..127
  const int alk = (tid & 3) * 8;       // chunk K offset
  const int sb1 = kb1 >> 5;            // segment boundaries in K-steps
  const int sb2 = kb2 >> 5;

  const u16 *a0p0, *a0p1, *a0p2, *a1p0, *a1p1, *a1p2;
  {
    const int m = m0 + ar0;
    const int r0 = idx0 ? idx0[m] : m;
    const int r1 = idx1 ? idx1[m] : m;
    const int r2 = idx2 ? idx2[m] : m;
    a0p0 = base0 + (size_t)r0 * st0 + alk;
    a0p1 = base1 + (size_t)r1 * st1 + alk;
    a0p2 = base2 + (size_t)r2 * st2 + alk;
  }
  {
    const int m = m0 + ar1;
    const int r0 = idx0 ? idx0[m] : m;
    const int r1 = idx1 ? idx1[m] : m;
    const int r2 = idx2 ? idx2[m] : m;
    a1p0 = base0 + (size_t)r0 * st0 + alk;
    a1p1 = base1 + (size_t)r1 * st1 + alk;
    a1p2 = base2 + (size_t)r2 * st2 + alk;
  }
  // B tile: 64x32 halves = 256 chunks; thread t owns chunk t.
  const u16* wrow = Wb + (size_t)(n0 + ar0) * Kpad + alk;

  // Wave-uniform segment select (s, sb1, sb2 scalar).
  auto aptr = [&](int s, const u16* p0, const u16* p1, const u16* p2) -> const u16* {
    return (s < sb1) ? (p0 + ((size_t)s << 5))
         : (s < sb2) ? (p1 + ((size_t)(s - sb1) << 5))
                     : (p2 + ((size_t)(s - sb2) << 5));
  };

  v8f acc0 = {}, acc1 = {}, acc2 = {}, acc3 = {};
  const int nsteps = Kpad >> 5;

  // Shared MMA stage: 2 A-frag + 8 B-frag ds_load_b128, 4 WMMAs.
  auto do_mma = [&](int buf) {
    const u16* As = At[buf];
    const u16* Bs = Bt[buf];
    union Frag { uint4 q[2]; v16bf v; };

    // A fragment (16x32): lane L<16 row L holds K {0..15},
    // lane L+16 holds K {16..31} (matches B).
    Frag fa;
    const int arow = wave * 16 + (lane & 15);
    const int akb  = (lane >> 4) * 16;
    fa.q[0] = *(const uint4*)&As[arow * 32 + akb];
    fa.q[1] = *(const uint4*)&As[arow * 32 + akb + 8];

    // B fragment (32x16): lane L holds column L%16, K (L/16)*16 + 0..15.
    Frag fb0, fb1, fb2, fb3;
    const int bb = (lane & 15) * 32 + (lane >> 4) * 16;
    fb0.q[0] = *(const uint4*)&Bs[bb];
    fb0.q[1] = *(const uint4*)&Bs[bb + 8];
    fb1.q[0] = *(const uint4*)&Bs[bb + 16 * 32];
    fb1.q[1] = *(const uint4*)&Bs[bb + 16 * 32 + 8];
    fb2.q[0] = *(const uint4*)&Bs[bb + 32 * 32];
    fb2.q[1] = *(const uint4*)&Bs[bb + 32 * 32 + 8];
    fb3.q[0] = *(const uint4*)&Bs[bb + 48 * 32];
    fb3.q[1] = *(const uint4*)&Bs[bb + 48 * 32 + 8];

    acc0 = __builtin_amdgcn_wmma_f32_16x16x32_bf16(false, fa.v, false, fb0.v,
                                                   (short)0, acc0, false, false);
    acc1 = __builtin_amdgcn_wmma_f32_16x16x32_bf16(false, fa.v, false, fb1.v,
                                                   (short)0, acc1, false, false);
    acc2 = __builtin_amdgcn_wmma_f32_16x16x32_bf16(false, fa.v, false, fb2.v,
                                                   (short)0, acc2, false, false);
    acc3 = __builtin_amdgcn_wmma_f32_16x16x32_bf16(false, fa.v, false, fb3.v,
                                                   (short)0, acc3, false, false);
  };

#if USE_ASYNC_LDS
  // ---- async global->LDS staging (ASYNCcnt) ------------------------------
  auto issue_tile = [&](int s, int buf) {
    const u16* pa0 = aptr(s, a0p0, a0p1, a0p2);
    const u16* pa1 = aptr(s, a1p0, a1p1, a1p2);
    const u16* pw  = wrow + ((size_t)s << 5);
    __builtin_amdgcn_global_load_async_to_lds_b128(
        to_g4(pa0), to_l4(&At[buf][ar0 * 32 + alk]), 0, 0);
    __builtin_amdgcn_global_load_async_to_lds_b128(
        to_g4(pa1), to_l4(&At[buf][ar1 * 32 + alk]), 0, 0);
    __builtin_amdgcn_global_load_async_to_lds_b128(
        to_g4(pw),  to_l4(&Bt[buf][ar0 * 32 + alk]), 0, 0);
    __builtin_prefetch(pw + 32, 0, 1);
  };
  issue_tile(0, 0);
  for (int s = 0; s < nsteps; ++s) {
    const int buf = s & 1;
    wait_async0();            // own wave's tile-s copies complete
    __syncthreads();          // -> whole tile s visible to all waves
    if (s + 1 < nsteps) issue_tile(s + 1, buf ^ 1);   // overlap with MMA
    do_mma(buf);
    // store(s+2) can only happen after sync(s+1), which post-dates every
    // thread's compute(s): double buffer needs no second barrier.
  }
#else
  // ---- fallback: pipelined global->reg->ds_store -------------------------
  uint4 av0 = *(const uint4*)aptr(0, a0p0, a0p1, a0p2);
  uint4 av1 = *(const uint4*)aptr(0, a1p0, a1p1, a1p2);
  uint4 wv  = *(const uint4*)wrow;
  for (int s = 0; s < nsteps; ++s) {
    const int buf = s & 1;
    *(uint4*)&At[buf][ar0 * 32 + alk] = av0;
    *(uint4*)&At[buf][ar1 * 32 + alk] = av1;
    *(uint4*)&Bt[buf][ar0 * 32 + alk] = wv;
    if (s + 1 < nsteps) {
      av0 = *(const uint4*)aptr(s + 1, a0p0, a0p1, a0p2);
      av1 = *(const uint4*)aptr(s + 1, a1p0, a1p1, a1p2);
      wv  = *(const uint4*)(wrow + ((size_t)(s + 1) << 5));
      __builtin_prefetch(wrow + ((size_t)(s + 2) << 5), 0, 1);
    }
    __syncthreads();
    do_mma(buf);
  }
#endif

  // Epilogue. C layout: VGPR r, lanes 0-15 -> row r, lanes 16-31 -> row r+8.
  const int mbase = m0 + wave * 16 + (lane >> 4) * 8;
  const int ncl   = n0 + (lane & 15);
  v8f* accs[4] = { &acc0, &acc1, &acc2, &acc3 };
#pragma unroll
  for (int g = 0; g < 4; ++g) {
    const int nc = ncl + g * 16;
    const float bb = bias[nc] * LRC;
    v8f& a = *accs[g];
#pragma unroll
    for (int r = 0; r < 8; ++r) {
      float y = a[r] * gain + bb;
      y = (y > 0.f ? y : 0.2f * y) * SQRT2C;
      out[(size_t)(mbase + r) * DDIM + nc] = f2b(y);
    }
  }
}

// ---------------------------------------------------------------------------
// Prep / aggregation / epilogue kernels
// ---------------------------------------------------------------------------
__global__ void zn_kernel(const float* __restrict__ z, float* __restrict__ zn) {
  __shared__ float red[256];
  const int b = blockIdx.x, t = threadIdx.x;
  const float a0 = z[b * 512 + t];
  const float a1 = z[b * 512 + 256 + t];
  red[t] = a0 * a0 + a1 * a1;
  __syncthreads();
  for (int s = 128; s > 0; s >>= 1) {
    if (t < s) red[t] += red[t + s];
    __syncthreads();
  }
  const float r = rsqrtf(red[0] / 512.f + 1e-8f);
  zn[b * 512 + t]       = a0 * r;
  zn[b * 512 + 256 + t] = a1 * r;
}

// x0 row i = [ zn[i%64] (512) | look_at[i] (3) | zero pad to 544 ]
__global__ void build_x(const float* __restrict__ zn, const float* __restrict__ la,
                        u16* __restrict__ xb) {
  const int idx = blockIdx.x * blockDim.x + threadIdx.x;
  if (idx >= MNODE * 544) return;
  const int i = idx / 544;
  const int c = idx - i * 544;
  float v = 0.f;
  if (c < 512)      v = zn[(i & 63) * 512 + c];            // jnp.tile -> row i%B
  else if (c < 515) v = la[i * 3 + (c - 512)];
  xb[idx] = f2b(v);
}

// edge_attr = [rel(3), dist(1)] padded to 32 bf16
__global__ void build_ea(const float* __restrict__ la, const int* __restrict__ src,
                         const int* __restrict__ dst, u16* __restrict__ eab) {
  const int e = blockIdx.x * blockDim.x + threadIdx.x;
  if (e >= EDGES) return;
  const int s = src[e], d = dst[e];
  const float rx = la[d * 3 + 0] - la[s * 3 + 0];
  const float ry = la[d * 3 + 1] - la[s * 3 + 1];
  const float rz = la[d * 3 + 2] - la[s * 3 + 2];
  const float dist = sqrtf(rx * rx + ry * ry + rz * rz);
  u16* p = eab + (size_t)e * 32;
  p[0] = f2b(rx); p[1] = f2b(ry); p[2] = f2b(rz); p[3] = f2b(dist);
#pragma unroll
  for (int c = 4; c < 32; ++c) p[c] = 0;
}

// Repack fp32 weight (512 x Korig) into bf16 (512 x Kpad) with up to 3
// (dstCol, srcCol, len) segments matching the padded A K-space.
__global__ void convw(const float* __restrict__ w, u16* __restrict__ wb,
                      int Korig, int Kpad,
                      int d0, int s0, int l0,
                      int d1, int s1, int l1,
                      int d2, int s2, int l2) {
  const int idx = blockIdx.x * blockDim.x + threadIdx.x;
  if (idx >= 512 * Kpad) return;
  const int n = idx / Kpad;
  const int k = idx - n * Kpad;
  float v = 0.f;
  if (k >= d0 && k < d0 + l0)            v = w[(size_t)n * Korig + s0 + (k - d0)];
  else if (l1 && k >= d1 && k < d1 + l1) v = w[(size_t)n * Korig + s1 + (k - d1)];
  else if (l2 && k >= d2 && k < d2 + l2) v = w[(size_t)n * Korig + s2 + (k - d2)];
  wb[idx] = f2b(v);
}

__global__ void zerof(float* __restrict__ p, int n) {
  const int idx = blockIdx.x * blockDim.x + threadIdx.x;
  if (idx < n) p[idx] = 0.f;
}

// segment-sum of ef over dst (+ in-degree counts)
__global__ void scatter_agg(const u16* __restrict__ ef, const int* __restrict__ dst,
                            float* __restrict__ sum, float* __restrict__ cnt) {
  const int idx = blockIdx.x * blockDim.x + threadIdx.x;
  if (idx >= EDGES * DDIM) return;
  const int e = idx >> 9;
  const int c = idx & 511;
  const int d = dst[e];
  atomicAdd(&sum[(size_t)d * DDIM + c], b2f(ef[idx]));
  if (c == 0) atomicAdd(&cnt[d], 1.0f);
}

// mode 0 (p0): aggb row = [mean(lookat tail)(3) | mean(ef)(512) | pad]
// mode 1 (p1): aggb row = [mean(ef)(512) | pad]
__global__ void agg_final(const float* __restrict__ sum, const float* __restrict__ cnt,
                          const float* __restrict__ la, u16* __restrict__ aggb,
                          int mode) {
  const int idx = blockIdx.x * blockDim.x + threadIdx.x;
  if (idx >= MNODE * 544) return;
  const int i = idx / 544;
  const int c = idx - i * 544;
  const float cc = cnt[i];
  const float denom = fmaxf(cc, 1.f);
  float v = 0.f;
  if (mode == 0) {
    if (c < 3)        v = (cc > 0.f) ? la[i * 3 + c] : 0.f;   // mean of repeated x[dst] tail
    else if (c < 515) v = sum[(size_t)i * DDIM + (c - 3)] / denom;
  } else {
    if (c < 512)      v = sum[(size_t)i * DDIM + c] / denom;
  }
  aggb[idx] = f2b(v);
}

// out[b, w, :] = x2[b*250, :]  (fp32 output)
__global__ void out_kernel(const u16* __restrict__ x2, float* __restrict__ out) {
  const int idx = blockIdx.x * blockDim.x + threadIdx.x;
  if (idx >= NB * NUMWS * DDIM) return;
  const int c  = idx & 511;
  const int bw = idx >> 9;
  const int b  = bw / NUMWS;
  out[idx] = b2f(x2[(size_t)(b * NV) * DDIM + c]);
}

// ---------------------------------------------------------------------------
// Launcher
// ---------------------------------------------------------------------------
extern "C" void kernel_launch(void* const* d_in, const int* in_sizes, int n_in,
                              void* d_out, int out_size, void* d_ws, size_t ws_size,
                              hipStream_t stream)
{
  (void)in_sizes; (void)n_in; (void)out_size; (void)ws_size;

  const float* z    = (const float*)d_in[0];
  const float* la   = (const float*)d_in[1];
  const int*   ei   = (const int*)d_in[2];
  const int*   esrc = ei;
  const int*   edst = ei + EDGES;

  const float* Wm[8] = { (const float*)d_in[3],  (const float*)d_in[5],
                         (const float*)d_in[7],  (const float*)d_in[9],
                         (const float*)d_in[11], (const float*)d_in[13],
                         (const float*)d_in[15], (const float*)d_in[17] };
  const float* Bv[8] = { (const float*)d_in[4],  (const float*)d_in[6],
                         (const float*)d_in[8],  (const float*)d_in[10],
                         (const float*)d_in[12], (const float*)d_in[14],
                         (const float*)d_in[16], (const float*)d_in[18] };

  // ---- workspace bump allocator -----------------------------------------
  char* wp = (char*)d_ws;
  auto take = [&](size_t bytes) -> char* {
    char* r = wp;
    wp += (bytes + 255) & ~(size_t)255;
    return r;
  };
  float* zn     = (float*)take((size_t)NB * 512 * 4);
  u16*   xb0    = (u16*)take((size_t)MNODE * 544 * 2);
  u16*   eab    = (u16*)take((size_t)EDGES * 32 * 2);
  const int Kpads[8] = {1120, 512, 1088, 512, 1536, 512, 1024, 512};
  const int Korig[8] = {1034, 512, 1030, 512, 1536, 512, 1024, 512};
  u16* Wb[8];
  for (int i = 0; i < 8; ++i) Wb[i] = (u16*)take((size_t)512 * Kpads[i] * 2);
  u16*   h_edge = (u16*)take((size_t)EDGES * DDIM * 2);
  u16*   ef0    = (u16*)take((size_t)EDGES * DDIM * 2);
  float* sum    = (float*)take((size_t)MNODE * DDIM * 4);
  float* cnt    = (float*)take((size_t)MNODE * 4);
  u16*   aggb   = (u16*)take((size_t)MNODE * 544 * 2);
  u16*   h_node = (u16*)take((size_t)MNODE * DDIM * 2);
  u16*   xb1    = (u16*)take((size_t)MNODE * DDIM * 2);
  u16*   xb2    = xb0;            // xb0 dead after p0 node fc1 -> reuse
  u16*   ef1    = ef0;            // ef0 dead after p1 edge fc1 -> reuse

  auto blocks = [](long n) { return dim3((unsigned)((n + 255) / 256)); };

  // ---- prep ---------------------------------------------------------------
  zn_kernel<<<NB, 256, 0, stream>>>(z, zn);
  build_x<<<blocks((long)MNODE * 544), 256, 0, stream>>>(zn, la, xb0);
  build_ea<<<blocks(EDGES), 256, 0, stream>>>(la, esrc, edst, eab);

  // ---- weight repack (segment maps mirror the padded A K-space) ----------
  struct SegMap { int d0, s0, l0, d1, s1, l1, d2, s2, l2; };
  const SegMap sm[8] = {
    {0, 0, 515,  544, 515, 515,  1088, 1030, 4},   // p0_ew0  K=1034 -> 1120
    {0, 0, 512,  0, 0, 0,        0, 0, 0},         // p0_ew1
    {0, 0, 515,  544, 515, 515,  0, 0, 0},         // p0_nw0  K=1030 -> 1088
    {0, 0, 512,  0, 0, 0,        0, 0, 0},         // p0_nw1
    {0, 0, 1536, 0, 0, 0,        0, 0, 0},         // p1_ew0
    {0, 0, 512,  0, 0, 0,        0, 0, 0},         // p1_ew1
    {0, 0, 1024, 0, 0, 0,        0, 0, 0},         // p1_nw0
    {0, 0, 512,  0, 0, 0,        0, 0, 0},         // p1_nw1
  };
  for (int i = 0; i < 8; ++i) {
    convw<<<blocks((long)512 * Kpads[i]), 256, 0, stream>>>(
        Wm[i], Wb[i], Korig[i], Kpads[i],
        sm[i].d0, sm[i].s0, sm[i].l0,
        sm[i].d1, sm[i].s1, sm[i].l1,
        sm[i].d2, sm[i].s2, sm[i].l2);
  }

  auto gemm = [&](const u16* b0, const int* i0, int st0,
                  const u16* b1, const int* i1, int st1,
                  const u16* b2, const int* i2, int st2,
                  int kb1, int kb2, int Kpad,
                  const u16* wb, const float* bias, float gain,
                  u16* out, int M) {
    gemm_fc<<<dim3((unsigned)(M / 128), 8), 256, 0, stream>>>(
        b0, i0, st0, b1, i1, st1, b2, i2, st2, kb1, kb2, Kpad, wb, bias, gain, out);
  };

  const float g1034 = LRC / sqrtf(1034.f);
  const float g512  = LRC / sqrtf(512.f);
  const float g1030 = LRC / sqrtf(1030.f);
  const float g1536 = LRC / sqrtf(1536.f);
  const float g1024 = LRC / sqrtf(1024.f);

  // ===== processor 0 =======================================================
  // edge fc1: ef_in = [x[src](544) | x[dst](544) | ea(32)]  K=1120
  gemm(xb0, esrc, 544, xb0, edst, 544, eab, nullptr, 32,
       544, 1088, 1120, Wb[0], Bv[0], g1034, h_edge, EDGES);
  // edge fc2: K=512
  gemm(h_edge, nullptr, 512, h_edge, nullptr, 512, h_edge, nullptr, 512,
       512, 512, 512, Wb[1], Bv[1], g512, ef0, EDGES);
  // segment mean of msg tail
  zerof<<<blocks((long)MNODE * DDIM), 256, 0, stream>>>(sum, MNODE * DDIM);
  zerof<<<blocks(MNODE), 256, 0, stream>>>(cnt, MNODE);
  scatter_agg<<<blocks((long)EDGES * DDIM), 256, 0, stream>>>(ef0, edst, sum, cnt);
  agg_final<<<blocks((long)MNODE * 544), 256, 0, stream>>>(sum, cnt, la, aggb, 0);
  // node fc1: nf_in = [x(544) | aggtail(544)]  K=1088
  gemm(xb0, nullptr, 544, aggb, nullptr, 544, aggb, nullptr, 544,
       544, 1088, 1088, Wb[2], Bv[2], g1030, h_node, MNODE);
  // node fc2
  gemm(h_node, nullptr, 512, h_node, nullptr, 512, h_node, nullptr, 512,
       512, 512, 512, Wb[3], Bv[3], g512, xb1, MNODE);

  // ===== processor 1 =======================================================
  // edge fc1: [x1[src](512) | x1[dst](512) | ef0(512)]  K=1536
  gemm(xb1, esrc, 512, xb1, edst, 512, ef0, nullptr, 512,
       512, 1024, 1536, Wb[4], Bv[4], g1536, h_edge, EDGES);
  // edge fc2 -> ef1 (reuses ef0 storage; ef0 consumed above)
  gemm(h_edge, nullptr, 512, h_edge, nullptr, 512, h_edge, nullptr, 512,
       512, 512, 512, Wb[5], Bv[5], g512, ef1, EDGES);
  // segment mean
  zerof<<<blocks((long)MNODE * DDIM), 256, 0, stream>>>(sum, MNODE * DDIM);
  zerof<<<blocks(MNODE), 256, 0, stream>>>(cnt, MNODE);
  scatter_agg<<<blocks((long)EDGES * DDIM), 256, 0, stream>>>(ef1, edst, sum, cnt);
  agg_final<<<blocks((long)MNODE * 544), 256, 0, stream>>>(sum, cnt, la, aggb, 1);
  // node fc1: [x1(512) | aggef(512, stride 544)]  K=1024
  gemm(xb1, nullptr, 512, aggb, nullptr, 544, aggb, nullptr, 544,
       512, 1024, 1024, Wb[6], Bv[6], g1024, h_node, MNODE);
  // node fc2 -> final node features (reuses xb0 storage)
  gemm(h_node, nullptr, 512, h_node, nullptr, 512, h_node, nullptr, 512,
       512, 512, 512, Wb[7], Bv[7], g512, xb2, MNODE);

  // ===== output: tile x[::250] to (64, 14, 512) fp32 =======================
  out_kernel<<<blocks((long)NB * NUMWS * DDIM), 256, 0, stream>>>(xb2, (float*)d_out);
}
